// SDEParallelFeatureHead_17420387352899
// MI455X (gfx1250) — matrix operand order repaired and checked by reading, not compiled, not run on hardware
//
#include <hip/hip_runtime.h>

// N=64, H=8, T=256, B=16, S=4, K=16, L=64, TOTAL=2560
#define Nn 64
#define Hh 8
#define Tt 256
#define Bb 16
#define Ss 4
#define Ll 64
#define Kk 16
#define TOTALF 2560
#define EPSV 1e-8f

typedef __attribute__((ext_vector_type(16))) _Float16 v16h;
typedef __attribute__((ext_vector_type(8)))  float    v8f;

__device__ __forceinline__ float waveMax(float v) {
#pragma unroll
  for (int o = 16; o > 0; o >>= 1) v = fmaxf(v, __shfl_xor(v, o, 32));
  return v;
}
__device__ __forceinline__ float waveSum(float v) {
#pragma unroll
  for (int o = 16; o > 0; o >>= 1) v += __shfl_xor(v, o, 32);
  return v;
}

// ---------------------------------------------------------------------------
// Kernel 1: fused softmax + head-average + all segment statistics.
// One wave32 per (b, s, n). Reads the 512MB input exactly once.
// Writes: node features -> feat[b][s*640 + n*9 + f]
//         edge features -> edge[(b*S+s)*4096 + n*64+c][4]
//         edge scores   -> score[(b*S+s)*4096 + n*64+c]
// ---------------------------------------------------------------------------
__global__ __launch_bounds__(256) void fused_stats_kernel(
    const float* __restrict__ sat, float* __restrict__ feat,
    float* __restrict__ edge, float* __restrict__ score) {
  const int waveInBlk = threadIdx.x >> 5;
  const int lane = threadIdx.x & 31;
  const int gw = blockIdx.x * 8 + waveInBlk;       // 0 .. 4095
  const int b = gw >> 8;                            // /(S*N)
  const int s = (gw >> 6) & 3;
  const int n = gw & 63;

  // per-lane edge accumulators (2 columns per lane: 2*lane, 2*lane+1)
  float sumP0 = 0.f, sumP1 = 0.f;
  float sumD0 = 0.f, sumD1 = 0.f;
  float sumD20 = 0.f, sumD21 = 0.f;
  float prev0 = 0.f, prev1 = 0.f;

  // wave-uniform scalar stats
  float entSum = 0.f, entSumSq = 0.f, entMax = -1e30f, entMin = 1e30f;
  float entFirst = 0.f, entLast = 0.f;
  float diagSum = 0.f, diagSumSq = 0.f;
  float diffSum = 0.f, diffSumSq = 0.f, diffMax = -1e30f;

  const size_t rowStrideH = (size_t)Nn * Nn;               // per head
  const size_t rowStrideT = (size_t)Hh * Nn * Nn;          // per timestep

  for (int lt = 0; lt < Ll; ++lt) {
    const int t = s * Ll + lt;
    const size_t base0 =
        (((size_t)b * Tt + t) * Hh) * Nn * Nn + (size_t)n * Nn;

    float avg0 = 0.f, avg1 = 0.f;
#pragma unroll
    for (int h = 0; h < Hh; ++h) {
      const float2 v =
          *(const float2*)(sat + base0 + (size_t)h * rowStrideH + lane * 2);
      float m = waveMax(fmaxf(v.x, v.y));
      float e0 = __expf(v.x - m);
      float e1 = __expf(v.y - m);
      float ssum = waveSum(e0 + e1);
      float inv = __frcp_rn(ssum);
      avg0 += e0 * inv;
      avg1 += e1 * inv;
    }
    avg0 *= 0.125f;
    avg1 *= 0.125f;

    // entropy over the row (Pc = max(P, EPS); -sum Pc*log(Pc))
    float pc0 = fmaxf(avg0, EPSV), pc1 = fmaxf(avg1, EPSV);
    float ent = waveSum(-(pc0 * __logf(pc0) + pc1 * __logf(pc1)));
    entSum += ent;
    entSumSq += ent * ent;
    entMax = fmaxf(entMax, ent);
    entMin = fminf(entMin, ent);
    if (lt == 0) entFirst = ent;
    entLast = ent;

    // diagonal element P[n][n]  (n is wave-uniform)
    float pick = (n & 1) ? avg1 : avg0;
    float dval = __shfl(pick, n >> 1, 32);
    diagSum += dval;
    diagSumSq += dval * dval;

    if (lt > 0) {
      float d0 = avg0 - prev0, d1 = avg1 - prev1;
      float diff = waveSum(fabsf(d0) + fabsf(d1));
      diffSum += diff;
      diffSumSq += diff * diff;
      diffMax = fmaxf(diffMax, diff);
      sumD0 += d0;  sumD1 += d1;
      sumD20 += d0 * d0;  sumD21 += d1 * d1;
    }
    prev0 = avg0;
    prev1 = avg1;
    sumP0 += avg0;
    sumP1 += avg1;
  }

  // ---- node features (uniform across wave; lane 0 writes) ----
  if (lane == 0) {
    float ent_mean = entSum * (1.f / 64.f);
    float ent_var = (entSumSq - entSum * entSum * (1.f / 64.f)) * (1.f / 63.f);
    float ent_std = sqrtf(fmaxf(ent_var, 0.f));
    float ent_range = entMax - entMin;
    float ent_slope = (entLast - entFirst) * (1.f / 63.f);
    float var_mean = diffSum * (1.f / 63.f);
    float var_var = (diffSumSq - diffSum * diffSum * (1.f / 63.f)) * (1.f / 62.f);
    float var_std = sqrtf(fmaxf(var_var, 0.f));
    float diag_mean = diagSum * (1.f / 64.f);
    float diag_var = (diagSumSq - diagSum * diagSum * (1.f / 64.f)) * (1.f / 63.f);
    float diag_std = sqrtf(fmaxf(diag_var, 0.f));

    float* nf = feat + (size_t)b * TOTALF + s * 640 + n * 9;
    nf[0] = ent_mean; nf[1] = ent_std; nf[2] = ent_range; nf[3] = ent_slope;
    nf[4] = var_mean; nf[5] = var_std; nf[6] = diffMax;
    nf[7] = diag_mean; nf[8] = diag_std;
  }

  // ---- edge features (2 edges per lane) ----
  const size_t ebase = ((size_t)(b * Ss + s)) * 4096 + (size_t)n * 64;
#pragma unroll
  for (int half = 0; half < 2; ++half) {
    int c = 2 * lane + half;
    float sp = half ? sumP1 : sumP0;
    float sd = half ? sumD1 : sumD0;
    float sd2 = half ? sumD21 : sumD20;
    float pem = sp * (1.f / 64.f);
    float dpm = sd * (1.f / 63.f);
    float dvar = (sd2 - sd * sd * (1.f / 63.f)) * (1.f / 62.f);
    float dstd = sqrtf(fmaxf(dvar, 0.f));
    float* ef = edge + (ebase + c) * 4;
    ef[0] = pem;
    ef[1] = dstd;
    ef[2] = fmaxf(dpm, 0.f);
    ef[3] = fmaxf(-dpm, 0.f);
    score[ebase + c] = (c == n) ? 0.f : fabsf(dpm);
  }
}

// ---------------------------------------------------------------------------
// Kernel 2: top-16 edge selection per (b,s) + gather into feat.
// ---------------------------------------------------------------------------
__global__ __launch_bounds__(256) void topk_gather_kernel(
    const float* __restrict__ edge, const float* __restrict__ score,
    float* __restrict__ feat) {
  __shared__ float sc[4096];
  __shared__ float rv[256];
  __shared__ int ri[256];
  __shared__ int sel[16];
  const int bs = blockIdx.x;  // 0..63
  const int tid = threadIdx.x;
  const float* sp = score + (size_t)bs * 4096;
  for (int i = tid; i < 4096; i += 256) sc[i] = sp[i];
  __syncthreads();

  for (int k = 0; k < Kk; ++k) {
    float best = -1.f;
    int bi = 0x7fffffff;
    for (int i = tid; i < 4096; i += 256) {
      float v = sc[i];
      if (v > best || (v == best && i < bi)) { best = v; bi = i; }
    }
    rv[tid] = best;
    ri[tid] = bi;
    __syncthreads();
    for (int off = 128; off > 0; off >>= 1) {
      if (tid < off) {
        if (rv[tid + off] > rv[tid] ||
            (rv[tid + off] == rv[tid] && ri[tid + off] < ri[tid])) {
          rv[tid] = rv[tid + off];
          ri[tid] = ri[tid + off];
        }
      }
      __syncthreads();
    }
    if (tid == 0) {
      sel[k] = ri[0];
      sc[ri[0]] = -2.f;
    }
    __syncthreads();
  }

  if (tid < Kk * 4) {
    int k = tid >> 2, c = tid & 3;
    int b = bs >> 2, s = bs & 3;
    feat[(size_t)b * TOTALF + s * 640 + 576 + k * 4 + c] =
        edge[((size_t)bs * 4096 + sel[k]) * 4 + c];
  }
}

// ---------------------------------------------------------------------------
// Kernel 3: LayerNorm per batch row -> f16 features for the WMMA head.
// ---------------------------------------------------------------------------
__global__ __launch_bounds__(256) void layernorm_kernel(
    const float* __restrict__ feat, const float* __restrict__ gamma,
    const float* __restrict__ beta, _Float16* __restrict__ h16) {
  __shared__ float red[256];
  const int b = blockIdx.x, tid = threadIdx.x;
  const float* f = feat + (size_t)b * TOTALF;

  float s = 0.f;
  for (int i = tid; i < TOTALF; i += 256) s += f[i];
  red[tid] = s;
  __syncthreads();
  for (int off = 128; off > 0; off >>= 1) {
    if (tid < off) red[tid] += red[tid + off];
    __syncthreads();
  }
  float mu = red[0] * (1.f / TOTALF);
  __syncthreads();

  float s2 = 0.f;
  for (int i = tid; i < TOTALF; i += 256) {
    float d = f[i] - mu;
    s2 += d * d;
  }
  red[tid] = s2;
  __syncthreads();
  for (int off = 128; off > 0; off >>= 1) {
    if (tid < off) red[tid] += red[tid + off];
    __syncthreads();
  }
  float rstd = rsqrtf(red[0] * (1.f / TOTALF) + 1e-5f);

  for (int i = tid; i < TOTALF; i += 256)
    h16[(size_t)b * TOTALF + i] =
        (_Float16)((f[i] - mu) * rstd * gamma[i] + beta[i]);
}

// ---------------------------------------------------------------------------
// Kernel 4: GEMM1 [16,2560] x [2560,128] + bias + ReLU via v_wmma_f32_16x16x32_f16
// 8 waves, one 16-column tile each; 80 K-steps of 32.
// A-frag element e (lane = half*16+m): k = (e>=8?16:0) + half*8 + ((e>>1)&3)*2 + (e&1)
// B-frag element e (lane = half*16+nn): k = (e>=8?16:0) + half*8 + (e&7)
// C/D vgpr r: row = half*8 + r, col = nn.
// ---------------------------------------------------------------------------
__global__ __launch_bounds__(256) void gemm1_wmma_kernel(
    const _Float16* __restrict__ h16, const float* __restrict__ W1,
    const float* __restrict__ b1, _Float16* __restrict__ hidden) {
  const int w = threadIdx.x >> 5;       // column tile 0..7
  const int lane = threadIdx.x & 31;
  const int half = lane >> 4;
  const int m = lane & 15;              // A row
  const int nn = lane & 15;             // B/D column within tile
  v8f acc = {};

  for (int k0 = 0; k0 < TOTALF; k0 += 32) {
    v16h a, bf;
#pragma unroll
    for (int e = 0; e < 16; ++e) {
      const int j = e >> 1;
      const int ka = ((j >= 4) ? 16 : 0) + half * 8 + (j & 3) * 2 + (e & 1);
      a[e] = h16[(size_t)m * TOTALF + k0 + ka];
      const int kb = ((e >= 8) ? 16 : 0) + half * 8 + (e & 7);
      bf[e] = (_Float16)W1[(size_t)(k0 + kb) * 128 + w * 16 + nn];
    }
    acc = __builtin_amdgcn_wmma_f32_16x16x32_f16(false, a, false, bf,
                                                 (short)0, acc, false, false);
  }

#pragma unroll
  for (int r = 0; r < 8; ++r) {
    const int row = half * 8 + r;
    const int col = w * 16 + nn;
    float v = acc[r] + b1[col];
    v = v > 0.f ? v : 0.f;
    hidden[row * 128 + col] = (_Float16)v;
  }
}

// ---------------------------------------------------------------------------
// Kernel 5: GEMM2 [16,128] x [128,64] + bias + ReLU -> fp32 output [16,64]
// 4 waves, one 16-column tile each; 4 K-steps of 32.
// ---------------------------------------------------------------------------
__global__ __launch_bounds__(128) void gemm2_wmma_kernel(
    const _Float16* __restrict__ hidden, const float* __restrict__ W2,
    const float* __restrict__ b2, float* __restrict__ out) {
  const int w = threadIdx.x >> 5;       // column tile 0..3
  const int lane = threadIdx.x & 31;
  const int half = lane >> 4;
  const int m = lane & 15;
  const int nn = lane & 15;
  v8f acc = {};

#pragma unroll
  for (int k0 = 0; k0 < 128; k0 += 32) {
    v16h a, bf;
#pragma unroll
    for (int e = 0; e < 16; ++e) {
      const int j = e >> 1;
      const int ka = ((j >= 4) ? 16 : 0) + half * 8 + (j & 3) * 2 + (e & 1);
      a[e] = hidden[(size_t)m * 128 + k0 + ka];
      const int kb = ((e >= 8) ? 16 : 0) + half * 8 + (e & 7);
      bf[e] = (_Float16)W2[(size_t)(k0 + kb) * 64 + w * 16 + nn];
    }
    acc = __builtin_amdgcn_wmma_f32_16x16x32_f16(false, a, false, bf,
                                                 (short)0, acc, false, false);
  }

#pragma unroll
  for (int r = 0; r < 8; ++r) {
    const int row = half * 8 + r;
    const int col = w * 16 + nn;
    float v = acc[r] + b2[col];
    out[row * 64 + col] = v > 0.f ? v : 0.f;
  }
}

// ---------------------------------------------------------------------------
extern "C" void kernel_launch(void* const* d_in, const int* in_sizes, int n_in,
                              void* d_out, int out_size, void* d_ws,
                              size_t ws_size, hipStream_t stream) {
  const float* sat   = (const float*)d_in[0];
  const float* gamma = (const float*)d_in[1];
  const float* beta  = (const float*)d_in[2];
  const float* W1    = (const float*)d_in[3];
  const float* b1    = (const float*)d_in[4];
  const float* W2    = (const float*)d_in[5];
  const float* b2    = (const float*)d_in[6];

  char* ws = (char*)d_ws;
  // workspace layout (bytes):
  //   edge  : B*S*4096*4 floats = 4,194,304
  //   score : B*S*4096 floats   = 1,048,576
  //   feat  : B*2560 floats     =   163,840
  //   h16   : B*2560 f16        =    81,920
  //   hidden: 16*128 f16        =     4,096
  float*    edge   = (float*)(ws);
  float*    score  = (float*)(ws + 4194304);
  float*    feat   = (float*)(ws + 4194304 + 1048576);
  _Float16* h16    = (_Float16*)(ws + 4194304 + 1048576 + 163840);
  _Float16* hidden = (_Float16*)(ws + 4194304 + 1048576 + 163840 + 81920);

  // 1. fused softmax/head-avg/stats: 4096 rows, 8 waves per block
  fused_stats_kernel<<<512, 256, 0, stream>>>(sat, feat, edge, score);
  // 2. top-16 edges per (b,s)
  topk_gather_kernel<<<Bb * Ss, 256, 0, stream>>>(edge, score, feat);
  // 3. layernorm -> f16
  layernorm_kernel<<<Bb, 256, 0, stream>>>(feat, gamma, beta, h16);
  // 4. MLP head with WMMA
  gemm1_wmma_kernel<<<1, 256, 0, stream>>>(h16, W1, b1, hidden);
  gemm2_wmma_kernel<<<1, 128, 0, stream>>>(hidden, W2, b2, (float*)d_out);
}
